// Submodel_89970974916774
// MI455X (gfx1250) — compile-verified, matrix operand order checked by reading, not run on hardware
//
#include <hip/hip_runtime.h>
#include <hip/hip_bf16.h>

// ---------------------------------------------------------------------------
// MI455X (gfx1250) implementation.
//
// Cost model: sims = onehot @ onehot_b^T is 4096x4096x3500 (~117 GFLOP) and
// sim @ [x|mask|mask*means_rows|ones] is 4096x4096x1504 (~50 GFLOP); all other
// stages are O(N*M). Intermediate traffic (~160 MB, fp16 operands fit in the
// 192 MB L2) costs single-digit us at 23.3 TB/s, so the problem is bound by
// matrix-core throughput -> do both GEMMs with v_wmma_f32_16x16x32_f16
// (f16 inputs exact for W_x/one-hots, f32 accumulate).
//
// Softmax kept unnormalized: p = exp((s-max)/T), Z = row-sum via a ones
// column in GEMM2; res = (p@x)/(p@mask + 1e-4*Z), same denominator for the
// means_rows_sum term (the exp(-max/T) factor cancels). Temperature depends
// on the global row max (it scales, not shifts), so a two-pass softmax with
// fp32 sims materialized is required.
//
// LDS staging uses the CDNA5 async direct-to-LDS path
// (GLOBAL_LOAD_ASYNC_TO_LDS_B128, ASYNCcnt) when the toolchain exposes the
// builtin; otherwise falls back to global_load_b128 + ds_store_b128.
// ---------------------------------------------------------------------------

typedef __attribute__((ext_vector_type(16))) _Float16 v16h;
typedef __attribute__((ext_vector_type(8)))  _Float16 v8h;
typedef __attribute__((ext_vector_type(8)))  float    v8f;
typedef int v4i __attribute__((vector_size(16)));   // matches builtin param type

#define N_USERS 4096
#define M_X     500
#define M_G     50
#define KDIM    3520      // 2500 + 1000, padded to multiple of 32
#define N2      1536      // 1504 (x|mask|mask*mr|ones) padded to multiple of 128

#if defined(__gfx1250__) && __has_builtin(__builtin_amdgcn_global_load_async_to_lds_b128)
#define HAVE_ASYNC_LDS 1
#else
#define HAVE_ASYNC_LDS 0
#endif

union Frag { v16h v; v8h h[2]; };

// ---------------------------------------------------------------------------
// K1: build fp16 embedding matrices A (values) and B (one-hots), [4096,3520]
// ---------------------------------------------------------------------------
__global__ void embed_kernel(const int* __restrict__ xi, const int* __restrict__ gi,
                             const float* __restrict__ Wx, const float* __restrict__ Ox,
                             const float* __restrict__ Wg, const float* __restrict__ Og,
                             _Float16* __restrict__ A, _Float16* __restrict__ B) {
    const int row = blockIdx.x;
    const int* xr = xi + (size_t)row * M_X;
    const int* gr = gi + (size_t)row * M_G;
    _Float16* Ar = A + (size_t)row * KDIM;
    _Float16* Br = B + (size_t)row * KDIM;
    for (int u = threadIdx.x; u < M_X + M_G + 20; u += blockDim.x) {
        if (u < M_X) {
            const int v = xr[u];
            const int base = u * 5;
            #pragma unroll
            for (int d = 0; d < 5; ++d) {
                Ar[base + d] = (_Float16)Wx[v * 5 + d];
                Br[base + d] = (_Float16)Ox[v * 5 + d];
            }
        } else if (u < M_X + M_G) {
            const int j = u - M_X;
            const int g = gr[j];
            const int base = 2500 + j * 20;
            #pragma unroll
            for (int d = 0; d < 20; ++d) {
                Ar[base + d] = (_Float16)Wg[g * 20 + d];
                Br[base + d] = (_Float16)Og[g * 20 + d];
            }
        } else {                         // zero pad cols 3500..3519
            const int c = 3500 + (u - (M_X + M_G));
            Ar[c] = (_Float16)0.0f;
            Br[c] = (_Float16)0.0f;
        }
    }
}

// ---------------------------------------------------------------------------
// Row / column statistics
// ---------------------------------------------------------------------------
__global__ void rowstats_kernel(const int* __restrict__ xi, float* __restrict__ mrows) {
    __shared__ float ssum[256], scnt[256];
    const int row = blockIdx.x;
    float s = 0.f, c = 0.f;
    for (int m = threadIdx.x; m < M_X; m += 256) {
        const int v = xi[(size_t)row * M_X + m];
        s += (float)v;
        c += (v > 0) ? 1.f : 0.f;
    }
    ssum[threadIdx.x] = s; scnt[threadIdx.x] = c;
    __syncthreads();
    for (int st = 128; st > 0; st >>= 1) {
        if (threadIdx.x < st) { ssum[threadIdx.x] += ssum[threadIdx.x + st];
                                scnt[threadIdx.x] += scnt[threadIdx.x + st]; }
        __syncthreads();
    }
    if (threadIdx.x == 0) mrows[row] = ssum[0] / (scnt[0] + 1e-5f);
}

__global__ void colstats_kernel(const int* __restrict__ xi, float* __restrict__ mcols) {
    __shared__ float ssum[256], scnt[256];
    const int col = blockIdx.x;
    float s = 0.f, c = 0.f;
    for (int r = threadIdx.x; r < N_USERS; r += 256) {
        const int v = xi[(size_t)r * M_X + col];
        s += (float)v;
        c += (v > 0) ? 1.f : 0.f;
    }
    ssum[threadIdx.x] = s; scnt[threadIdx.x] = c;
    __syncthreads();
    for (int st = 128; st > 0; st >>= 1) {
        if (threadIdx.x < st) { ssum[threadIdx.x] += ssum[threadIdx.x + st];
                                scnt[threadIdx.x] += scnt[threadIdx.x + st]; }
        __syncthreads();
    }
    if (threadIdx.x == 0) mcols[col] = ssum[0] / (scnt[0] + 1e-5f);
}

__global__ void meanreduce_kernel(const float* __restrict__ mrows,
                                  const float* __restrict__ mcols,
                                  float* __restrict__ means2) {
    __shared__ float sm[256];
    float s = 0.f;
    for (int i = threadIdx.x; i < N_USERS; i += 256) s += mrows[i];
    sm[threadIdx.x] = s;
    __syncthreads();
    for (int st = 128; st > 0; st >>= 1) {
        if (threadIdx.x < st) sm[threadIdx.x] += sm[threadIdx.x + st];
        __syncthreads();
    }
    if (threadIdx.x == 0) means2[0] = sm[0] / (float)N_USERS;
    __syncthreads();
    s = 0.f;
    for (int i = threadIdx.x; i < M_X; i += 256) s += mcols[i];
    sm[threadIdx.x] = s;
    __syncthreads();
    for (int st = 128; st > 0; st >>= 1) {
        if (threadIdx.x < st) sm[threadIdx.x] += sm[threadIdx.x + st];
        __syncthreads();
    }
    if (threadIdx.x == 0) means2[1] = sm[0] / (float)M_X;
}

// ---------------------------------------------------------------------------
// WMMA GEMM: C[M,N] = A[M,K] * B[N,K]^T  (both row-major fp16, C fp32)
// Block tile 128x128, 8 wave32 waves, each wave 4x2 wmma accumulators.
// Requires M%128==0 (grid.y), N%128==0 (grid.x), K%32==0.
// ---------------------------------------------------------------------------
__global__ __launch_bounds__(256)
void gemm_abt_wmma(const _Float16* __restrict__ A, const _Float16* __restrict__ B,
                   float* __restrict__ C, int K,
                   int lda, int ldb, int ldc, int zeroDiag) {
    __shared__ __align__(16) _Float16 As[128][40];   // +8 halves pad vs banks
    __shared__ __align__(16) _Float16 Bs[128][40];

    const int tid  = threadIdx.x;
    const int lane = tid & 31;
    const int wave = tid >> 5;
    const int wm   = wave >> 2;        // 0..1  (64 rows each)
    const int wn   = wave & 3;         // 0..3  (32 cols each)
    const int bm   = blockIdx.y * 128;
    const int bn   = blockIdx.x * 128;

    const v8f vzero = {0.f, 0.f, 0.f, 0.f, 0.f, 0.f, 0.f, 0.f};
    v8f acc[4][2];
    #pragma unroll
    for (int i = 0; i < 4; ++i)
        #pragma unroll
        for (int j = 0; j < 2; ++j) acc[i][j] = vzero;

    // ISA 16-bit operand layout (CDNA5 ISA): lanes 0-15 hold K {0..7,16..23}
    // of row (lane&15); lanes 16-31 hold K {8..15,24..31}.
    const int mrow = lane & 15;
    const int hi8  = (lane >> 4) * 8;

    for (int k0 = 0; k0 < K; k0 += 32) {
        // ---- stage 128x32 fp16 tiles of A and B into LDS ----
        #pragma unroll
        for (int p = 0; p < 2; ++p) {
            const int idx = tid + p * 256;            // 0..511
            const int r   = idx >> 2;                 // 0..127
            const int c   = (idx & 3) * 8;            // 0,8,16,24
            const _Float16* gA = &A[(size_t)(bm + r) * lda + k0 + c];
            const _Float16* gB = &B[(size_t)(bn + r) * ldb + k0 + c];
#if HAVE_ASYNC_LDS
            // CDNA5 direct-to-LDS async copy (16B per lane, ASYNCcnt)
            __builtin_amdgcn_global_load_async_to_lds_b128(
                (__attribute__((address_space(1))) v4i*)gA,
                (__attribute__((address_space(3))) v4i*)&As[r][c], 0, 0);
            __builtin_amdgcn_global_load_async_to_lds_b128(
                (__attribute__((address_space(1))) v4i*)gB,
                (__attribute__((address_space(3))) v4i*)&Bs[r][c], 0, 0);
#else
            *(v8h*)&As[r][c] = *(const v8h*)gA;
            *(v8h*)&Bs[r][c] = *(const v8h*)gB;
#endif
            if (k0 + 32 < K) {   // prefetch next K tile -> global_prefetch_b8
                __builtin_prefetch(&A[(size_t)(bm + r) * lda + k0 + 32 + c], 0, 0);
                __builtin_prefetch(&B[(size_t)(bn + r) * ldb + k0 + 32 + c], 0, 0);
            }
        }
#if HAVE_ASYNC_LDS
        // Drain this wave's async LDS writes before the workgroup barrier.
#if __has_builtin(__builtin_amdgcn_s_wait_asynccnt)
        __builtin_amdgcn_s_wait_asynccnt(0);
#else
        asm volatile("s_wait_asynccnt 0" ::: "memory");
#endif
#endif
        __syncthreads();

        // ---- load fragments from LDS (two ds_load_b128 per fragment) ----
        Frag af[4], bf[2];
        #pragma unroll
        for (int mt = 0; mt < 4; ++mt) {
            const int r = wm * 64 + mt * 16 + mrow;
            af[mt].h[0] = *(const v8h*)&As[r][hi8];
            af[mt].h[1] = *(const v8h*)&As[r][hi8 + 16];
        }
        #pragma unroll
        for (int nt = 0; nt < 2; ++nt) {
            const int r = wn * 32 + nt * 16 + mrow;
            bf[nt].h[0] = *(const v8h*)&Bs[r][hi8];
            bf[nt].h[1] = *(const v8h*)&Bs[r][hi8 + 16];
        }

        // ---- 8 WMMA ops per wave per K step ----
        #pragma unroll
        for (int mt = 0; mt < 4; ++mt)
            #pragma unroll
            for (int nt = 0; nt < 2; ++nt)
                acc[mt][nt] = __builtin_amdgcn_wmma_f32_16x16x32_f16(
                    false, af[mt].v, false, bf[nt].v,
                    (short)0, acc[mt][nt], false, false);
        __syncthreads();
    }

    // ---- epilogue: C/D layout — VGPR r holds (M = r + 8*(lane>>4), N = lane&15)
    const int mOff = (lane >> 4) * 8;
    const int nIdx = lane & 15;
    #pragma unroll
    for (int mt = 0; mt < 4; ++mt)
        #pragma unroll
        for (int nt = 0; nt < 2; ++nt)
            #pragma unroll
            for (int r = 0; r < 8; ++r) {
                const int gm = bm + wm * 64 + mt * 16 + mOff + r;
                const int gn = bn + wn * 32 + nt * 16 + nIdx;
                float v = acc[mt][nt][r];
                if (zeroDiag && gm == gn) v = 0.f;
                C[(size_t)gm * ldc + gn] = v;
            }
}

// ---------------------------------------------------------------------------
// Row max of sims -> inverse temperature (diag is 0 so max >= 0)
// ---------------------------------------------------------------------------
__global__ void rowmax_kernel(const float* __restrict__ S,
                              const float* __restrict__ beta,
                              const float* __restrict__ gamma,
                              float* __restrict__ rmax, float* __restrict__ invT) {
    __shared__ float sm[256];
    const int row = blockIdx.x;
    const float* sr = S + (size_t)row * N_USERS;
    float m = -3.0e38f;
    for (int j = threadIdx.x; j < N_USERS; j += 256) m = fmaxf(m, sr[j]);
    sm[threadIdx.x] = m;
    __syncthreads();
    for (int st = 128; st > 0; st >>= 1) {
        if (threadIdx.x < st) sm[threadIdx.x] = fmaxf(sm[threadIdx.x], sm[threadIdx.x + st]);
        __syncthreads();
    }
    if (threadIdx.x == 0) {
        const float mx = sm[0];
        const float T  = powf(mx + 0.001f, gamma[0]) * beta[0];
        rmax[row] = mx;
        invT[row] = 1.0f / T;
    }
}

// ---------------------------------------------------------------------------
// Unnormalized shifted softmax numerator -> fp16
// ---------------------------------------------------------------------------
__global__ void exp_kernel(const float* __restrict__ S,
                           const float* __restrict__ rmax,
                           const float* __restrict__ invT,
                           _Float16* __restrict__ P) {
    const int row = blockIdx.x;
    const float m  = rmax[row];
    const float it = invT[row];
    const float* sr = S + (size_t)row * N_USERS;
    _Float16*    pr = P + (size_t)row * N_USERS;
    for (int j = threadIdx.x; j < N_USERS; j += 256)
        pr[j] = (_Float16)__expf((sr[j] - m) * it);
}

// ---------------------------------------------------------------------------
// Build B2^T [1536, 4096]: rows 0..499 = x, 500..999 = mask,
// 1000..1499 = mask * means_rows, 1500 = ones, 1501..1535 = zeros.
// ---------------------------------------------------------------------------
__global__ void b2t_kernel(const int* __restrict__ xi,
                           const float* __restrict__ mrows,
                           _Float16* __restrict__ B2) {
    const int r = blockIdx.x;
    _Float16* br = B2 + (size_t)r * N_USERS;
    if (r < 500) {
        for (int j = threadIdx.x; j < N_USERS; j += 256)
            br[j] = (_Float16)(float)xi[(size_t)j * M_X + r];
    } else if (r < 1000) {
        const int m = r - 500;
        for (int j = threadIdx.x; j < N_USERS; j += 256)
            br[j] = (_Float16)((xi[(size_t)j * M_X + m] > 0) ? 1.0f : 0.0f);
    } else if (r < 1500) {
        const int m = r - 1000;
        for (int j = threadIdx.x; j < N_USERS; j += 256)
            br[j] = (_Float16)((xi[(size_t)j * M_X + m] > 0) ? mrows[j] : 0.0f);
    } else if (r == 1500) {
        for (int j = threadIdx.x; j < N_USERS; j += 256) br[j] = (_Float16)1.0f;
    } else {
        for (int j = threadIdx.x; j < N_USERS; j += 256) br[j] = (_Float16)0.0f;
    }
}

// ---------------------------------------------------------------------------
// Final: res + MeanAdd correction, clamp(1,5)
// ---------------------------------------------------------------------------
__global__ void final_kernel(const float* __restrict__ O,
                             const float* __restrict__ mrows,
                             const float* __restrict__ mcols,
                             const float* __restrict__ means2,
                             const float* __restrict__ coef,
                             float* __restrict__ out) {
    const int row = blockIdx.x;
    const float* orow = O + (size_t)row * N2;
    const float Z   = orow[1500];
    const float mr  = mrows[row];
    const float mrm = means2[0];
    const float mcm = means2[1];
    const float c0 = coef[0], c1 = coef[1], c2 = coef[2];
    for (int m = threadIdx.x; m < M_X; m += 256) {
        const float D   = orow[500 + m] + 1e-4f * Z;   // p@mask + 1e-4 * Z
        const float res = orow[m] / D;                 // (p@x) / D
        const float mra = mr - orow[1000 + m] / D;     // means_rows - means_rows_sum
        const float add = c0 * mra + c1 * (mcols[m] - mcm) + c2 * (mr - mrm);
        float v = res + add;
        v = fminf(fmaxf(v, 1.0f), 5.0f);
        out[(size_t)row * M_X + m] = v;
    }
}

// ---------------------------------------------------------------------------
// Host launcher
// ---------------------------------------------------------------------------
extern "C" void kernel_launch(void* const* d_in, const int* in_sizes, int n_in,
                              void* d_out, int out_size, void* d_ws, size_t ws_size,
                              hipStream_t stream) {
    const int*   x_idx = (const int*)  d_in[0];
    const int*   g_idx = (const int*)  d_in[1];
    const float* W_x   = (const float*)d_in[2];
    const float* ohe_x = (const float*)d_in[3];
    const float* W_g   = (const float*)d_in[4];
    const float* ohe_g = (const float*)d_in[5];
    const float* beta  = (const float*)d_in[6];
    const float* gamma = (const float*)d_in[7];
    const float* coef  = (const float*)d_in[8];
    float* out = (float*)d_out;

    char* ws = (char*)d_ws;
    // Workspace layout (with aliasing; ~158.4 MB total):
    const size_t szAB = (size_t)N_USERS * KDIM * sizeof(_Float16);   // 28,835,840 B
    const size_t offA = 0;                       // A_half; later reused for O (fp32, 25.2 MB)
    const size_t offB = offA + szAB;             // B_half; later reused for B2t (12.6 MB)
    const size_t offS = offB + szAB;             // sims fp32: 64 MB
    const size_t offP = offS + (size_t)N_USERS * N_USERS * sizeof(float);     // P fp16: 32 MB
    const size_t offStats = offP + (size_t)N_USERS * N_USERS * sizeof(_Float16);

    _Float16* A_half = (_Float16*)(ws + offA);
    _Float16* B_half = (_Float16*)(ws + offB);
    float*    sims   = (float*)   (ws + offS);
    _Float16* P_half = (_Float16*)(ws + offP);
    float*    O_f32  = (float*)   (ws + offA);   // alias: A dead after GEMM1
    _Float16* B2t    = (_Float16*)(ws + offB);   // alias: B dead after GEMM1

    float* mrows  = (float*)(ws + offStats);     // [4096]
    float* mcols  = mrows + N_USERS;             // [512] (500 used)
    float* rmax   = mcols + 512;                 // [4096]
    float* invT   = rmax + N_USERS;              // [4096]
    float* means2 = invT + N_USERS;              // [2]

    // 1) fp16 embedding matrices
    embed_kernel<<<N_USERS, 256, 0, stream>>>(x_idx, g_idx, W_x, ohe_x, W_g, ohe_g,
                                              A_half, B_half);
    // 2) statistics
    rowstats_kernel<<<N_USERS, 256, 0, stream>>>(x_idx, mrows);
    colstats_kernel<<<M_X, 256, 0, stream>>>(x_idx, mcols);
    meanreduce_kernel<<<1, 256, 0, stream>>>(mrows, mcols, means2);

    // 3) GEMM1: sims = A * B^T, zero diagonal  (4096 x 4096 x 3520)
    gemm_abt_wmma<<<dim3(N_USERS / 128, N_USERS / 128), 256, 0, stream>>>
        (A_half, B_half, sims, KDIM, KDIM, KDIM, N_USERS, 1);

    // 4) row max -> inverse temperature
    rowmax_kernel<<<N_USERS, 256, 0, stream>>>(sims, beta, gamma, rmax, invT);

    // 5) P = fp16(exp((s - max) * invT))
    exp_kernel<<<N_USERS, 256, 0, stream>>>(sims, rmax, invT, P_half);

    // 6) B2^T (must follow GEMM1: aliases B_half region)
    b2t_kernel<<<N2, 256, 0, stream>>>(x_idx, mrows, B2t);

    // 7) GEMM2: O = P * B2t^T  (4096 x 1536 x 4096); O aliases A_half region
    gemm_abt_wmma<<<dim3(N2 / 128, N_USERS / 128), 256, 0, stream>>>
        (P_half, B2t, O_f32, N_USERS, N_USERS, N_USERS, N2, 0);

    // 8) epilogue
    final_kernel<<<N_USERS, 256, 0, stream>>>(O_f32, mrows, mcols, means2, coef, out);
}
